// PointsetAbstraction_6055903887569
// MI455X (gfx1250) — matrix operand order repaired and checked by reading, not compiled, not run on hardware
//
#include <hip/hip_runtime.h>
#include <hip/hip_bf16.h>

typedef __attribute__((ext_vector_type(16))) _Float16 v16h;
typedef __attribute__((ext_vector_type(8)))  float    v8f;

#define Bc 16
#define Nc 4096
#define Sc 1024

// ---------------------------------------------------------------------------
// 1. Transpose pos (B,3,N)->xyzT (B*N,3) f32  and feats (B,64,N)->featsT (B*N,64) f16
// ---------------------------------------------------------------------------
__global__ void transpose_inputs(const float* __restrict__ pos,
                                 const float* __restrict__ feats,
                                 float* __restrict__ xyzT,
                                 _Float16* __restrict__ featsT) {
  long t = (long)blockIdx.x * 256 + threadIdx.x;      // < B*N
  int b = (int)(t >> 12);
  int n = (int)(t & (Nc - 1));
  const float* pb = pos + (long)b * 3 * Nc;
  float* xo = xyzT + t * 3;
  xo[0] = pb[n];
  xo[1] = pb[Nc + n];
  xo[2] = pb[2 * Nc + n];
  const float* fb = feats + (long)b * 64 * Nc;
  _Float16* fo = featsT + t * 64;
#pragma unroll
  for (int c = 0; c < 64; ++c) fo[c] = (_Float16)fb[(long)c * Nc + n];
}

// ---------------------------------------------------------------------------
// 2. Farthest point sampling: 1 block (256 thr) per batch; coords in registers
// ---------------------------------------------------------------------------
__global__ void fps_kernel(const float* __restrict__ xyzT,
                           float* __restrict__ newXyz,
                           float* __restrict__ outPos) {
  const int b = blockIdx.x;
  const int tid = threadIdx.x;                        // 256 threads
  float px[16], py[16], pz[16], dist[16];
#pragma unroll
  for (int i = 0; i < 16; ++i) {
    int n = i * 256 + tid;
    const float* p = xyzT + ((long)b * Nc + n) * 3;
    px[i] = p[0]; py[i] = p[1]; pz[i] = p[2];
    dist[i] = 1e10f;
  }
  __shared__ int sCur;
  __shared__ float rv[256];
  __shared__ int ri[256];
  if (tid == 0) sCur = 0;
  __syncthreads();
  for (int s = 0; s < Sc; ++s) {
    int cur = sCur;
    const float* c = xyzT + ((long)b * Nc + cur) * 3;
    float cx = c[0], cy = c[1], cz = c[2];
    if (tid == 0) {
      long o = ((long)b * Sc + s) * 3;
      newXyz[o] = cx; newXyz[o + 1] = cy; newXyz[o + 2] = cz;
      long po = (long)b * 3 * Sc + s;
      outPos[po] = cx; outPos[po + Sc] = cy; outPos[po + 2 * Sc] = cz;
    }
    float best = -1.f;
    int bi = 0;
#pragma unroll
    for (int i = 0; i < 16; ++i) {
      float dx = px[i] - cx, dy = py[i] - cy, dz = pz[i] - cz;
      float d = dx * dx + dy * dy + dz * dz;
      float dd = fminf(dist[i], d);
      dist[i] = dd;
      if (dd > best) { best = dd; bi = i * 256 + tid; }
    }
    rv[tid] = best; ri[tid] = bi;
    __syncthreads();
    for (int off = 128; off > 0; off >>= 1) {
      if (tid < off) {
        float ov = rv[tid + off]; int oi = ri[tid + off];
        if (ov > rv[tid] || (ov == rv[tid] && oi < ri[tid])) { rv[tid] = ov; ri[tid] = oi; }
      }
      __syncthreads();
    }
    if (tid == 0) sCur = ri[0];
    __syncthreads();
  }
}

// ---------------------------------------------------------------------------
// 3. Ball query + gather: one wave32 per (b,s); ballot prefix-sum selection.
// ---------------------------------------------------------------------------
__global__ void ballquery_gather(const float* __restrict__ xyzT,
                                 const _Float16* __restrict__ featsT,
                                 const float* __restrict__ newXyz,
                                 _Float16* __restrict__ X0,
                                 float r2, int K) {
  const int tid = threadIdx.x;
  const int lane = tid & 31;
  const int wave = tid >> 5;
  const int gid = blockIdx.x * 8 + wave;              // b*S + s
  const int b = gid >> 10;
  const int s = gid & (Sc - 1);
  __shared__ int list[8][64];

  const float* ctr = newXyz + ((long)b * Sc + s) * 3;
  float cx = ctr[0], cy = ctr[1], cz = ctr[2];

  int have = 0, chunk = 0;
  while (have < K && chunk < Nc / 32) {
    int n = chunk * 32 + lane;
    const float* p = xyzT + ((long)b * Nc + n) * 3;
    float dx = p[0] - cx, dy = p[1] - cy, dz = p[2] - cz;
    int in = (dx * dx + dy * dy + dz * dz) <= r2 ? 1 : 0;
    unsigned mask = (unsigned)__ballot(in);
    int pos = have + __popc(mask & ((1u << lane) - 1u));
    if (in && pos < K) list[wave][pos] = n;
    have = min(have + __popc(mask), K);
    ++chunk;
  }
  int first = list[wave][0];                          // center itself always qualifies
  for (int j = have + lane; j < K; j += 32) list[wave][j] = first;

  long rowBase = (((long)gid) * K) * 128;
  for (int j = 0; j < K; ++j) {
    int idx = list[wave][j];
    const _Float16* fr = featsT + ((long)b * Nc + idx) * 64;
    _Float16* dst = X0 + rowBase + (long)j * 128;
    *(unsigned*)&dst[2 * lane] = *(const unsigned*)&fr[2 * lane];   // cols 0..63
    float v = 0.f;
    if (lane < 3) {
      v = xyzT[((long)b * Nc + idx) * 3 + lane] - ((lane == 0) ? cx : (lane == 1) ? cy : cz);
    }
    dst[64 + lane] = (_Float16)v;                                   // cols 64..95 (zero pad)
  }
}

// ---------------------------------------------------------------------------
// 4. WMMA GEMM + BN-stats, async double-buffered X tile.
//    If NORM: apply previous layer's BN scale/shift + ReLU on the A fragment
//    (folds the inter-layer normalization pass into this GEMM's load).
//    X tile staged via global_load_async_to_lds_b128 (ASYNCcnt, 2 buffers).
// ---------------------------------------------------------------------------
template <int NTILES, bool NORM>
__global__ void gemm_stats_wmma(const _Float16* __restrict__ X,
                                const _Float16* __restrict__ Wh,
                                const float* __restrict__ bias,
                                const float* __restrict__ ssIn,
                                _Float16* __restrict__ Hout,
                                float* __restrict__ stats, int Kpad) {
  constexpr int XS = 40;                               // Xs row stride (halves)
  __shared__ __align__(16) _Float16 Ws[128 * 136];
  __shared__ __align__(16) _Float16 Xs[2][128 * XS];
  __shared__ float sSum[128], sSqs[128];
  __shared__ float sScale[128], sShift[128];

  const int tid = threadIdx.x;
  const int cout = NTILES * 16;
  const int wstride = Kpad + 8;                        // keeps 16B row alignment
  const int segs = Kpad >> 3;

  for (int i = tid; i < cout * segs; i += 256) {
    int r = i / segs, s0 = i % segs;
    *(uint4*)&Ws[r * wstride + s0 * 8] = *(const uint4*)&Wh[r * Kpad + s0 * 8];
  }
  if (tid < 128) {
    sSum[tid] = 0.f; sSqs[tid] = 0.f;
    if (NORM) { sScale[tid] = ssIn[tid]; sShift[tid] = ssIn[128 + tid]; }
  }

  const long mBase = (long)blockIdx.x * 128;
  v8f acc[NTILES];
#pragma unroll
  for (int t = 0; t < NTILES; ++t) acc[t] = (v8f){0.f, 0.f, 0.f, 0.f, 0.f, 0.f, 0.f, 0.f};

  const int lane = tid & 31, wave = tid >> 5;
  const int halfSel = lane >> 4, rl = lane & 15;
  const int koffA = halfSel * 8;                       // A: K {0-7,16-23}/{8-15,24-31}
  const int koffB = halfSel * 8;                       // B: K {0-7,16-23}/{8-15,24-31}

  // --- async X-tile staging: each thread moves one 32B row-segment per chunk
  const int ldRow = tid >> 1, ldPart = (tid & 1) * 16;
  const unsigned gByteBase = (unsigned)((((long)blockIdx.x * 128 + ldRow) * 128 + ldPart) * 2);
  auto issue = [&](int kc, int buf) {
    unsigned ldsOff = (unsigned)(uintptr_t)(const void*)&Xs[buf][ldRow * XS + ldPart];
    unsigned gOff = gByteBase + (unsigned)(kc * 64);
    asm volatile("global_load_async_to_lds_b128 %0, %1, %2\n\t"
                 "global_load_async_to_lds_b128 %0, %1, %2 offset:16"
                 :: "v"(ldsOff), "v"(gOff), "s"(X) : "memory");
  };

  const int kchunks = Kpad >> 5;
  issue(0, 0);
  for (int kc = 0; kc < kchunks; ++kc) {
    const int buf = kc & 1;
    if (kc + 1 < kchunks) {
      issue(kc + 1, buf ^ 1);
      asm volatile("s_wait_asynccnt 2" ::: "memory");  // chunk kc landed (in-order)
    } else {
      asm volatile("s_wait_asynccnt 0" ::: "memory");
    }
    __syncthreads();                                   // all waves' data visible

    const _Float16* arow = &Xs[buf][(wave * 16 + rl) * XS];
    union { uint4 u[2]; v16h v; _Float16 h[16]; } af;
    af.u[0] = *(const uint4*)(arow + koffA);
    af.u[1] = *(const uint4*)(arow + koffA + 16);
    if (NORM) {                                        // fused BN + ReLU of prev layer
#pragma unroll
      for (int i = 0; i < 16; ++i) {
        int kk = kc * 32 + koffA + ((i < 8) ? i : i + 8);
        float v = (float)af.h[i] * sScale[kk] + sShift[kk];
        af.h[i] = (_Float16)fmaxf(v, 0.f);
      }
    }
#pragma unroll
    for (int t = 0; t < NTILES; ++t) {
      const _Float16* brow = &Ws[(t * 16 + rl) * wstride + kc * 32 + koffB];
      union { uint4 u[2]; v16h v; } bf;
      bf.u[0] = *(const uint4*)(brow);
      bf.u[1] = *(const uint4*)(brow + 16);
      acc[t] = __builtin_amdgcn_wmma_f32_16x16x32_f16(
          false, af.v, false, bf.v, (short)0, acc[t], false, false);
    }
    __syncthreads();                                   // protect buffer reuse
  }

#pragma unroll
  for (int t = 0; t < NTILES; ++t) {
    int n = t * 16 + rl;
    float bv = bias[n];
    float ls = 0.f, lq = 0.f;
#pragma unroll
    for (int vg = 0; vg < 8; ++vg) {
      float h = acc[t][vg] + bv;
      long m = mBase + wave * 16 + halfSel * 8 + vg;
      Hout[m * 128 + n] = (_Float16)h;
      ls += h; lq += h * h;
    }
    atomicAdd(&sSum[n], ls);                           // ds_add_f32
    atomicAdd(&sSqs[n], lq);
  }
  __syncthreads();
  if (tid < cout) {
    atomicAdd(&stats[tid], sSum[tid]);
    atomicAdd(&stats[128 + tid], sSqs[tid]);
  }
}

// ---------------------------------------------------------------------------
// 5. Small helpers
// ---------------------------------------------------------------------------
__global__ void prep_w(const float* __restrict__ W, _Float16* __restrict__ Wh,
                       int cout, int cin, int Kpad) {
  int g = blockIdx.x * 256 + threadIdx.x;
  if (g >= cout * Kpad) return;
  int r = g / Kpad, c = g % Kpad;
  Wh[g] = (c < cin) ? (_Float16)W[r * cin + c] : (_Float16)0.f;
}

__global__ void zero_stats(float* stats) { stats[threadIdx.x] = 0.f; }

__global__ void stats_finalize(const float* __restrict__ stats,
                               const float* __restrict__ gamma,
                               const float* __restrict__ beta,
                               float* __restrict__ ss, int cout, float inv) {
  int c = threadIdx.x;
  if (c >= cout) return;
  float mu = stats[c] * inv;
  float var = stats[128 + c] * inv - mu * mu;
  float sc = gamma[c] * rsqrtf(var + 1e-5f);
  ss[c] = sc;
  ss[128 + c] = beta[c] - mu * sc;
}

// Final layer: BN + ReLU + max over k, transposed store into new_feats (B,256,S)
__global__ void norm_max_out(const _Float16* __restrict__ H, const float* __restrict__ ss,
                             float* __restrict__ outFeat, int K, int chOff) {
  int bs = blockIdx.x;                                 // b*S+s
  int c = threadIdx.x;                                 // 128
  float sc = ss[c], sh = ss[128 + c];
  const _Float16* base = H + (long)bs * K * 128 + c;
  float m = -3.4e38f;
  for (int j = 0; j < K; ++j) m = fmaxf(m, (float)base[(long)j * 128] * sc + sh);
  m = fmaxf(m, 0.f);
  int b = bs >> 10, s = bs & (Sc - 1);
  outFeat[(long)b * 256 * Sc + (long)(chOff + c) * Sc + s] = m;
}

// ---------------------------------------------------------------------------
extern "C" void kernel_launch(void* const* d_in, const int* in_sizes, int n_in,
                              void* d_out, int out_size, void* d_ws, size_t ws_size,
                              hipStream_t stream) {
  const float* pos = (const float*)d_in[0];
  const float* feats = (const float*)d_in[1];
  auto leaf = [&](int br, int ly, int comp) -> const float* {
    return (const float*)d_in[2 + br * 12 + ly * 4 + comp];
  };

  // workspace layout (~546 MB)
  char* ws = (char*)d_ws;
  size_t off = 0;
  auto alloc = [&](size_t bytes) { size_t r = off; off = (off + bytes + 255) & ~(size_t)255; return r; };
  float*    xyzT   = (float*)(ws + alloc((size_t)Bc * Nc * 3 * 4));
  _Float16* featsT = (_Float16*)(ws + alloc((size_t)Bc * Nc * 64 * 2));
  float*    newXyz = (float*)(ws + alloc((size_t)Bc * Sc * 3 * 4));
  _Float16* Wh     = (_Float16*)(ws + alloc((size_t)128 * 128 * 2));
  float*    stats  = (float*)(ws + alloc(256 * 4));
  float*    ss     = (float*)(ws + alloc(256 * 4));
  const size_t XBUF = (size_t)Bc * Sc * 64 * 128 * 2;  // worst case M=1,048,576 rows
  _Float16* Xa = (_Float16*)(ws + alloc(XBUF));
  _Float16* Xb = (_Float16*)(ws + alloc(XBUF));

  float* outPos  = (float*)d_out;                      // (B,3,S)
  float* outFeat = outPos + (size_t)Bc * 3 * Sc;       // (B,256,S)

  transpose_inputs<<<(Bc * Nc) / 256, 256, 0, stream>>>(pos, feats, xyzT, featsT);
  fps_kernel<<<Bc, 256, 0, stream>>>(xyzT, newXyz, outPos);

  const int   Ks[2]       = {32, 64};
  const float Rs[2]       = {0.1f, 0.2f};
  const int   couts[2][3] = {{64, 64, 128}, {64, 96, 128}};

  for (int br = 0; br < 2; ++br) {
    const int K = Ks[br];
    const long M = (long)Bc * Sc * K;                  // multiple of 128
    ballquery_gather<<<(Bc * Sc) / 8, 256, 0, stream>>>(xyzT, featsT, newXyz, Xa,
                                                        Rs[br] * Rs[br], K);
    int cin = 67;
    _Float16* bufs[2] = {Xa, Xb};
    for (int ly = 0; ly < 3; ++ly) {
      const int cout = couts[br][ly];
      const int Kpad = (cin + 31) & ~31;               // 96,64,64 / 96,64,96
      const float* W  = leaf(br, ly, 0);
      const float* bi = leaf(br, ly, 1);
      const float* ga = leaf(br, ly, 2);
      const float* be = leaf(br, ly, 3);
      _Float16* Xsrc = bufs[ly & 1];
      _Float16* Xdst = bufs[(ly + 1) & 1];

      prep_w<<<(cout * Kpad + 255) / 256, 256, 0, stream>>>(W, Wh, cout, cin, Kpad);
      zero_stats<<<1, 256, 0, stream>>>(stats);

      const int grid = (int)(M / 128);
      if (ly == 0) {                                   // raw input, no fused BN
        gemm_stats_wmma<4, false><<<grid, 256, 0, stream>>>(Xsrc, Wh, bi, ss, Xdst, stats, Kpad);
      } else {
        switch (cout / 16) {                           // fused BN+ReLU of prev layer
          case 4: gemm_stats_wmma<4, true><<<grid, 256, 0, stream>>>(Xsrc, Wh, bi, ss, Xdst, stats, Kpad); break;
          case 6: gemm_stats_wmma<6, true><<<grid, 256, 0, stream>>>(Xsrc, Wh, bi, ss, Xdst, stats, Kpad); break;
          default: gemm_stats_wmma<8, true><<<grid, 256, 0, stream>>>(Xsrc, Wh, bi, ss, Xdst, stats, Kpad); break;
        }
      }
      stats_finalize<<<1, 128, 0, stream>>>(stats, ga, be, ss, cout, 1.0f / (float)M);

      if (ly == 2) {
        norm_max_out<<<Bc * Sc, 128, 0, stream>>>(Xdst, ss, outFeat, K, br * 128);
      }
      cin = cout;
    }
  }
}
